// DogDetector_40535901339909
// MI455X (gfx1250) — compile-verified, matrix operand order checked by reading, not run on hardware
//
#include <hip/hip_runtime.h>
#include <hip/hip_bf16.h>

typedef __attribute__((ext_vector_type(16))) _Float16 v16h;
typedef __attribute__((ext_vector_type(8)))  _Float16 v8h;
typedef __attribute__((ext_vector_type(8)))  float    v8f;

// ---------------- workspace layout (all _Float16) ----------------
#define N_WLAT (256 * 512)
#define N_WSM  (9 * 256 * 256)
#define N_WBB  (9 * 48 * 256)
#define N_WCL  (9 * 16 * 256)
#define OFF_WLAT 0
#define OFF_WSM  (OFF_WLAT + N_WLAT)
#define OFF_WC1  (OFF_WSM + N_WSM)
#define OFF_WC2  (OFF_WC1 + N_WSM)
#define OFF_WBB  (OFF_WC2 + N_WSM)
#define OFF_WCL  (OFF_WBB + N_WBB)

// ---------------- weight repack fp32 -> f16, per-tap [t][oc][ic] ----------------
__global__ void dd_convert_weights(const float* __restrict__ wlat,
                                   const float* __restrict__ wsm,
                                   const float* __restrict__ wc1,
                                   const float* __restrict__ wc2,
                                   const float* __restrict__ wbb,
                                   const float* __restrict__ wcl,
                                   _Float16* __restrict__ ws) {
  const int stride = gridDim.x * blockDim.x;
  const int g = blockIdx.x * blockDim.x + threadIdx.x;
  for (int i = g; i < N_WLAT; i += stride)              // [oc][ic] natural
    ws[OFF_WLAT + i] = (_Float16)wlat[i];
  for (int i = g; i < N_WSM; i += stride) {             // src [oc][ic][3][3]
    const int t = i >> 16, r = i & 65535, oc = r >> 8, ic = r & 255;
    const int s = oc * 2304 + ic * 9 + t;
    ws[OFF_WSM + i] = (_Float16)wsm[s];
    ws[OFF_WC1 + i] = (_Float16)wc1[s];
    ws[OFF_WC2 + i] = (_Float16)wc2[s];
  }
  for (int i = g; i < N_WBB; i += stride) {             // pad oc 36->48
    const int t = i / (48 * 256), r = i % (48 * 256), oc = r >> 8, ic = r & 255;
    ws[OFF_WBB + i] = (oc < 36) ? (_Float16)wbb[oc * 2304 + ic * 9 + t] : (_Float16)0.f;
  }
  for (int i = g; i < N_WCL; i += stride) {             // pad oc 9->16
    const int t = i / (16 * 256), r = i % (16 * 256), oc = r >> 8, ic = r & 255;
    ws[OFF_WCL + i] = (oc < 9) ? (_Float16)wcl[oc * 2304 + ic * 9 + t] : (_Float16)0.f;
  }
}

// ---------------- WMMA helpers ----------------
// 16-bit A/B fragment: row = lane&15; lanes 0-15 carry K[0..7],K[16..23],
// lanes 16-31 carry K[8..15],K[24..31] (per CDNA5 ISA 7.12.2). base points at
// row start + k0; rows are contiguous in K, so both chunks are 16B loads.
static __device__ __forceinline__ v16h dd_load_frag(const _Float16* base, int half) {
  union { v16h v; v8h h[2]; } u;
  u.h[0] = *(const v8h*)(base + (half << 3));
  u.h[1] = *(const v8h*)(base + 16 + (half << 3));
  return u.v;
}

static __device__ __forceinline__ v8f dd_wmma(v16h a, v16h b, v8f c) {
  return __builtin_amdgcn_wmma_f32_16x16x32_f16(false, a, false, b, (short)0, c, false, false);
}

// ---------------- conv building blocks (one image in LDS) ----------------
// Padding is data-driven: invalid lanes read a dedicated all-zero LDS row,
// so B loads are unconditional (no EXEC toggling between WMMAs).

// 1x1 conv: D[oc][pos] = W[oc][512] x X[512][pos], X in LDS as [pos][512], row 49 = zeros
static __device__ __forceinline__ void dd_conv1x1(const _Float16* sX, _Float16* dst,
                                                  const _Float16* W, const float* bias,
                                                  int wid, int lane) {
  const int half = lane >> 4, rr = lane & 15;
  for (int tile = wid; tile < 64; tile += 8) {
    const int oc0 = (tile >> 2) << 4;
    const int pos = ((tile & 3) << 4) + rr;
    const int prow = (pos < 49) ? pos : 49;  // zero row
    const _Float16* arow = W + (size_t)(oc0 + rr) * 512;
    const _Float16* brow = sX + prow * 512;
    v8f acc = {};
    for (int k0 = 0; k0 < 512; k0 += 32) {
      v16h a = dd_load_frag(arow + k0, half);
      v16h b = dd_load_frag(brow + k0, half);
      acc = dd_wmma(a, b, acc);
    }
    const int ocb = oc0 + (half << 3);
    v8h sv;
#pragma unroll
    for (int j = 0; j < 8; ++j) sv[j] = (_Float16)(acc[j] + bias[ocb + j]);
    *(v8h*)(dst + pos * 256 + ocb) = sv;  // dst has 65 rows; rows 49..63 unread, 64 = zeros
  }
}

// 3x3 conv 256->256, pad=1, per-tap GEMMs; src/dst LDS [pos][256], row 64 = zeros
static __device__ __forceinline__ void dd_conv3x3(const _Float16* src, _Float16* dst,
                                                  const _Float16* W, const float* bias,
                                                  bool relu, int wid, int lane) {
  const int half = lane >> 4, rr = lane & 15;
  for (int tile = wid; tile < 64; tile += 8) {
    const int oc0 = (tile >> 2) << 4;
    const int pos = ((tile & 3) << 4) + rr;
    const int py = pos / 7, px = pos % 7;
    const bool pin = pos < 49;
    v8f acc = {};
    for (int t = 0; t < 9; ++t) {
      const int sy = py + t / 3 - 1, sx = px + t % 3 - 1;
      const bool v = pin && sy >= 0 && sy < 7 && sx >= 0 && sx < 7;
      const int ps = v ? (sy * 7 + sx) : 64;  // zero row
      const _Float16* arow = W + (size_t)t * 65536 + (size_t)(oc0 + rr) * 256;
      const _Float16* brow = src + ps * 256;
      for (int k0 = 0; k0 < 256; k0 += 32) {
        v16h a = dd_load_frag(arow + k0, half);
        v16h b = dd_load_frag(brow + k0, half);
        acc = dd_wmma(a, b, acc);
      }
    }
    const int ocb = oc0 + (half << 3);
    v8h sv;
#pragma unroll
    for (int j = 0; j < 8; ++j) {
      float val = acc[j] + bias[ocb + j];
      if (relu) val = fmaxf(val, 0.f);
      sv[j] = (_Float16)val;
    }
    *(v8h*)(dst + pos * 256 + ocb) = sv;
  }
}

// bbox (36ch, padded 48 -> 12 tiles) + conf (9ch, padded 16 -> 4 tiles)
static __device__ __forceinline__ void dd_heads(const _Float16* src,
                                                float* sBBox, float* sScore,
                                                const _Float16* Wbb, const float* bbb,
                                                const _Float16* Wcl, const float* bcl,
                                                int wid, int lane) {
  const int half = lane >> 4, rr = lane & 15;
  for (int tile = wid; tile < 16; tile += 8) {  // uniform per wave
    const bool isB = tile < 12;
    const int oc0 = isB ? ((tile >> 2) << 4) : 0;
    const int pos = ((isB ? (tile & 3) : (tile - 12)) << 4) + rr;
    const int py = pos / 7, px = pos % 7;
    const bool pin = pos < 49;
    v8f acc = {};
    for (int t = 0; t < 9; ++t) {
      const int sy = py + t / 3 - 1, sx = px + t % 3 - 1;
      const bool v = pin && sy >= 0 && sy < 7 && sx >= 0 && sx < 7;
      const int ps = v ? (sy * 7 + sx) : 64;  // zero row
      const _Float16* arow =
          (isB ? (Wbb + (size_t)t * (48 * 256)) : (Wcl + (size_t)t * (16 * 256))) +
          (size_t)(oc0 + rr) * 256;
      const _Float16* brow = src + ps * 256;
      for (int k0 = 0; k0 < 256; k0 += 32) {
        v16h a = dd_load_frag(arow + k0, half);
        v16h b = dd_load_frag(brow + k0, half);
        acc = dd_wmma(a, b, acc);
      }
    }
    const int ocb = oc0 + (half << 3);
    if (pin) {
      if (isB) {
#pragma unroll
        for (int j = 0; j < 8; ++j) {
          const int oc = ocb + j;
          if (oc < 36) sBBox[pos * 36 + oc] = acc[j] + bbb[oc];
        }
      } else {
#pragma unroll
        for (int j = 0; j < 8; ++j) {
          const int oc = ocb + j;
          if (oc < 9) {
            const float s = acc[j] + bcl[oc];
            sScore[pos * 9 + oc] = 1.f / (1.f + expf(-s));
          }
        }
      }
    }
  }
}

// ---------------- fused per-image detector ----------------
__global__ __launch_bounds__(256) void dd_main(
    const float* __restrict__ x, const _Float16* __restrict__ ws,
    const float* __restrict__ b_lat, const float* __restrict__ b_sm,
    const float* __restrict__ b_c1, const float* __restrict__ b_c2,
    const float* __restrict__ b_bb, const float* __restrict__ b_cl,
    const float* __restrict__ w1, const float* __restrict__ b1,
    const float* __restrict__ w2, const float* __restrict__ b2,
    float* __restrict__ out, int B) {
  __shared__ _Float16 sX[50 * 512];   // x transposed [pos][ic], f16; row 49 = zeros
  __shared__ _Float16 sF0[65 * 256];  // ping; row 64 = zeros
  __shared__ _Float16 sF1[65 * 256];  // pong; row 64 = zeros
  __shared__ float sBBox[49 * 36];
  __shared__ float sScore[441];
  __shared__ float sBoxes[441 * 4];
  __shared__ float sS[441];
  __shared__ float sSort[512];
  __shared__ float sGap[256];
  __shared__ float sHid[128];
  __shared__ float sRedV[256];
  __shared__ int   sRedI[256];
  __shared__ float sKeepS[100];
  __shared__ float sKeepB[400];
  __shared__ float sCount;

  const int tid = threadIdx.x;
  const int b = blockIdx.x;
  const int lane = tid & 31;
  const int wid = tid >> 5;

  // ---- stage 0: load + transpose x into LDS as f16; init zero rows ----
  const float* xb = x + (size_t)b * (512 * 49);
  for (int i = tid; i < 512 * 49; i += 256) {
    const int c = i / 49, p = i % 49;
    sX[p * 512 + c] = (_Float16)xb[i];
  }
  for (int i = tid; i < 512; i += 256) sX[49 * 512 + i] = (_Float16)0.f;
  if (tid < 256) {
    sF0[64 * 256 + tid] = (_Float16)0.f;
    sF1[64 * 256 + tid] = (_Float16)0.f;
  }
  __syncthreads();

  // ---- conv chain (all WMMA) ----
  dd_conv1x1(sX, sF0, ws + OFF_WLAT, b_lat, wid, lane);            // lat
  __syncthreads();
  dd_conv3x3(sF0, sF1, ws + OFF_WSM, b_sm, false, wid, lane);      // feat
  __syncthreads();
  dd_conv3x3(sF1, sF0, ws + OFF_WC1, b_c1, true, wid, lane);       // h1 (relu)
  __syncthreads();
  dd_conv3x3(sF0, sF1, ws + OFF_WC2, b_c2, true, wid, lane);       // h2 (relu)
  __syncthreads();
  dd_heads(sF1, sBBox, sScore, ws + OFF_WBB, b_bb, ws + OFF_WCL, b_cl, wid, lane);
  __syncthreads();

  // ---- GAP over positions (h2) ----
  {
    float s = 0.f;
    for (int p = 0; p < 49; ++p) s += (float)sF1[p * 256 + tid];
    sGap[tid] = s * (1.f / 49.f);
  }

  // ---- anchor decode + clip ----
  for (int a = tid; a < 441; a += 256) {
    const int pos = a / 9, j = a % 9;
    const float sc3[3] = {0.5f, 1.f, 2.f};
    const float s_ = sc3[j / 3], r_ = sc3[j % 3];
    const float sq = sqrtf(r_);
    const float aw = s_ * sq, ah = s_ / sq;
    const float acx = ((pos % 7) + 0.5f) * (1.f / 7.f);
    const float acy = ((pos / 7) + 0.5f) * (1.f / 7.f);
    const float tx = sBBox[pos * 36 + 4 * j + 0];
    const float ty = sBBox[pos * 36 + 4 * j + 1];
    const float tw = sBBox[pos * 36 + 4 * j + 2];
    const float th = sBBox[pos * 36 + 4 * j + 3];
    const float cx = tx * aw + acx, cy = ty * ah + acy;
    const float bw = expf(tw) * aw, bh = expf(th) * ah;
    sBoxes[a * 4 + 0] = fminf(fmaxf(cx - bw * 0.5f, 0.f), 1.f);
    sBoxes[a * 4 + 1] = fminf(fmaxf(cy - bh * 0.5f, 0.f), 1.f);
    sBoxes[a * 4 + 2] = fminf(fmaxf(cx + bw * 0.5f, 0.f), 1.f);
    sBoxes[a * 4 + 3] = fminf(fmaxf(cy + bh * 0.5f, 0.f), 1.f);
  }
  __syncthreads();

  // ---- count MLP ----
  if (tid < 128) {
    float h = b1[tid];
    for (int c = 0; c < 256; ++c) h += w1[tid * 256 + c] * sGap[c];
    sHid[tid] = fmaxf(h, 0.f);
  }
  __syncthreads();
  if (tid == 0) {
    float s = b2[0];
    for (int j = 0; j < 128; ++j) s += w2[j] * sHid[j];
    sCount = fmaxf(s, 0.f);
  }
  __syncthreads();

  // ---- bitonic sort (descending) for k-th score ----
  for (int i = tid; i < 512; i += 256) sSort[i] = (i < 441) ? sScore[i] : -1e30f;
  __syncthreads();
  for (int k = 2; k <= 512; k <<= 1) {
    for (int j = k >> 1; j > 0; j >>= 1) {
      for (int i = tid; i < 512; i += 256) {
        const int l = i ^ j;
        if (l > i) {
          const float a0 = sSort[i], a1 = sSort[l];
          const bool desc = ((i & k) == 0);
          if (desc ? (a0 < a1) : (a0 > a1)) { sSort[i] = a1; sSort[l] = a0; }
        }
      }
      __syncthreads();
    }
  }

  // ---- adaptive threshold ----
  const float cnt = sCount;
  int tca = (int)floorf(cnt + 0.5f);
  tca = min(max(tca, 1), 100);
  const float thr = fminf(fmaxf(sSort[tca - 1] * 0.95f, 0.15f), 0.5f);
  for (int a = tid; a < 441; a += 256) {
    const float v = sScore[a];
    sS[a] = (v > thr) ? v : -1.f;
  }
  __syncthreads();

  // ---- greedy NMS, 100 iterations ----
  for (int it = 0; it < 100; ++it) {
    float bv = -2e30f; int bi = 0;
    for (int a = tid; a < 441; a += 256) {
      const float v = sS[a];
      if (v > bv) { bv = v; bi = a; }
    }
    sRedV[tid] = bv; sRedI[tid] = bi;
    __syncthreads();
    for (int s2 = 128; s2 > 0; s2 >>= 1) {
      if (tid < s2) {
        const float ov = sRedV[tid + s2]; const int oi = sRedI[tid + s2];
        const float cv = sRedV[tid];      const int ci = sRedI[tid];
        if (ov > cv || (ov == cv && oi < ci)) { sRedV[tid] = ov; sRedI[tid] = oi; }
      }
      __syncthreads();
    }
    const int idx = sRedI[0];
    const float sc = sRedV[0];
    const float q0 = sBoxes[idx * 4 + 0], q1 = sBoxes[idx * 4 + 1];
    const float q2 = sBoxes[idx * 4 + 2], q3 = sBoxes[idx * 4 + 3];
    if (tid == 0) {
      sKeepS[it] = sc;
      sKeepB[it * 4 + 0] = q0; sKeepB[it * 4 + 1] = q1;
      sKeepB[it * 4 + 2] = q2; sKeepB[it * 4 + 3] = q3;
    }
    const float areaQ = (q2 - q0) * (q3 - q1);
    for (int a = tid; a < 441; a += 256) {
      const float p0 = sBoxes[a * 4 + 0], p1 = sBoxes[a * 4 + 1];
      const float p2 = sBoxes[a * 4 + 2], p3 = sBoxes[a * 4 + 3];
      const float ix0 = fmaxf(q0, p0), iy0 = fmaxf(q1, p1);
      const float ix1 = fminf(q2, p2), iy1 = fminf(q3, p3);
      const float inter = fmaxf(ix1 - ix0, 0.f) * fmaxf(iy1 - iy0, 0.f);
      const float areaP = (p2 - p0) * (p3 - p1);
      const float iou = inter / fmaxf(areaQ + areaP - inter, 1e-9f);
      if (iou > 0.45f) sS[a] = -1.f;
    }
    __syncthreads();
  }

  // ---- emit: kb | ks | valid | count (concatenated, fp32) ----
  int tct = (int)floorf(cnt + 0.5f);
  tct = min(max(tct, 0), 100);
  const size_t ksOff = (size_t)B * 400;
  const size_t vOff  = ksOff + (size_t)B * 100;
  const size_t cOff  = vOff + (size_t)B * 100;
  for (int i = tid; i < 100; i += 256) {
    const float scv = sKeepS[i];
    const bool val = (scv > 0.f) && (i < tct);
    float* kb = out + (size_t)b * 400 + (size_t)i * 4;
#pragma unroll
    for (int c = 0; c < 4; ++c) kb[c] = val ? sKeepB[i * 4 + c] : 0.f;
    out[ksOff + (size_t)b * 100 + i] = val ? scv : 0.f;
    out[vOff + (size_t)b * 100 + i] = val ? 1.f : 0.f;
  }
  if (tid == 0) out[cOff + b] = cnt;
}

extern "C" void kernel_launch(void* const* d_in, const int* in_sizes, int n_in,
                              void* d_out, int out_size, void* d_ws, size_t ws_size,
                              hipStream_t stream) {
  const float* x     = (const float*)d_in[0];
  const float* w_lat = (const float*)d_in[1];
  const float* b_lat = (const float*)d_in[2];
  const float* w_sm  = (const float*)d_in[3];
  const float* b_sm  = (const float*)d_in[4];
  const float* w_c1  = (const float*)d_in[5];
  const float* b_c1  = (const float*)d_in[6];
  const float* w_c2  = (const float*)d_in[7];
  const float* b_c2  = (const float*)d_in[8];
  const float* w_bb  = (const float*)d_in[9];
  const float* b_bb  = (const float*)d_in[10];
  const float* w_cl  = (const float*)d_in[11];
  const float* b_cl  = (const float*)d_in[12];
  const float* w1    = (const float*)d_in[13];
  const float* b1    = (const float*)d_in[14];
  const float* w2    = (const float*)d_in[15];
  const float* b2    = (const float*)d_in[16];
  _Float16* ws = (_Float16*)d_ws;
  const int B = in_sizes[0] / (512 * 49);

  dd_convert_weights<<<512, 256, 0, stream>>>(w_lat, w_sm, w_c1, w_c2, w_bb, w_cl, ws);
  dd_main<<<B, 256, 0, stream>>>(x, ws, b_lat, b_sm, b_c1, b_c2, b_bb, b_cl,
                                 w1, b1, w2, b2, (float*)d_out, B);
}
